// TransformerDQN_68109591380106
// MI455X (gfx1250) — matrix operand order repaired and checked by reading, not compiled, not run on hardware
//
#include <hip/hip_runtime.h>

typedef __attribute__((ext_vector_type(16))) _Float16 v16h;
typedef __attribute__((ext_vector_type(8)))  _Float16 v8h;
typedef __attribute__((ext_vector_type(8)))  float    v8f;
typedef __attribute__((ext_vector_type(4)))  unsigned int v4u;
typedef __attribute__((ext_vector_type(8)))  int v8i;
typedef __attribute__((ext_vector_type(4)))  int v4i;

// ---------------------------------------------------------------------------
// f32 -> f16 conversion
// ---------------------------------------------------------------------------
__global__ void k_cvt(const float* __restrict__ src, _Float16* __restrict__ dst, int n) {
    int i = blockIdx.x * blockDim.x + threadIdx.x;
    if (i < n) dst[i] = (_Float16)src[i];
}

// ---------------------------------------------------------------------------
// Generic GEMM: C[M,N] = act(A[M,K](lda) @ W[N,K]^T + bias), f16 in/out,
// f32 accumulate via v_wmma_f32_16x16x32_f16.
//
// Block = 128 threads (4 waves). All 4 waves share one 64-column B panel:
// the whole 64xK panel (<=64KB) is staged into LDS ONCE per block by the
// Tensor Data Mover (tensor_load_to_lds + s_wait_tensorcnt), then each wave
// reads its B fragments as contiguous 32B ds_load_b128s. All four fragments
// are loaded into distinct registers before the WMMA group so the LDS loads
// clause together behind one s_wait_dscnt instead of four serialized waits.
//
// Requires: M%64==0, N%64==0, K%32==0, lda%8==0.  Dynamic LDS: K*64 halfs.
// ---------------------------------------------------------------------------
__global__ void k_gemm(const _Float16* __restrict__ A, const _Float16* __restrict__ W,
                       const float* __restrict__ bias, _Float16* __restrict__ C,
                       int M, int N, int K, int lda, int act) {
    extern __shared__ _Float16 Bs[];  // [64][K] rows n-nBase, packed
    const int lane = threadIdx.x & 31;
    const int wave = threadIdx.x >> 5;
    const int mTile = blockIdx.x * 4 + wave;
    const int m0 = mTile * 16;
    const int nBase = blockIdx.y * 64;
    const int half = lane >> 4;
    const int lm = lane & 15;
    const int am = m0 + lm;

    // ---- stage B panel (64 rows x K halfs) into LDS via TDM ----
#if __has_builtin(__builtin_amdgcn_tensor_load_to_lds)
    if (wave == 0) {
        unsigned lds = (unsigned)(size_t)Bs;  // flat LDS aperture: low 32 bits = LDS offset
        unsigned long long ga = (unsigned long long)(size_t)(W + (size_t)nBase * K);
        unsigned td0 = (unsigned)(K >> 2);    // row length in 8-byte units
        const unsigned rows = 64;
        // D# group 0: count=1 | lds_addr | global_addr(57b) | type=2
        v4u g0;
        g0[0] = 1u;
        g0[1] = lds;
        g0[2] = (unsigned)ga;
        g0[3] = (unsigned)((ga >> 32) & 0x01ffffffu) | (2u << 30);
        // D# group 1: data_size=3(8B); tensor_dim0=td0; tensor_dim1=rows;
        //             tile_dim0=td0; tile_dim1=rows; tile_dim2=0;
        //             tensor_dim0_stride=td0; tensor_dim1_stride=0
        v8i g1;
        g1[0] = (int)(3u << 16);
        g1[1] = (int)((td0 & 0xffffu) << 16);
        g1[2] = (int)((td0 >> 16) | ((rows & 0xffffu) << 16));
        g1[3] = (int)((rows >> 16) | ((td0 & 0xffffu) << 16));
        g1[4] = (int)(rows & 0xffffu);
        g1[5] = (int)td0;
        g1[6] = 0;
        g1[7] = 0;
        v4i gz4 = {0, 0, 0, 0};
        v8i gz8 = {0, 0, 0, 0, 0, 0, 0, 0};
        __builtin_amdgcn_tensor_load_to_lds(g0, g1, gz4, gz4, gz8, 0);
        __builtin_amdgcn_s_wait_tensorcnt((short)0);
    }
#else
    {   // fallback: cooperative copy
        const v8h* src = (const v8h*)(W + (size_t)nBase * K);
        v8h* dst = (v8h*)Bs;
        int chunks = (64 * K) / 8;
        for (int idx = threadIdx.x; idx < chunks; idx += 128) dst[idx] = src[idx];
    }
#endif
    __syncthreads();

    if (m0 < M) {
        v8f zero = {};
        v8f acc[4];
#pragma unroll
        for (int t = 0; t < 4; ++t) acc[t] = zero;

        for (int kk = 0; kk < K; kk += 32) {
            const _Float16* ap = A + (size_t)am * lda + kk + half * 16;
            v8h alo = *(const v8h*)(ap);
            v8h ahi = *(const v8h*)(ap + 8);
            v16h a;
#pragma unroll
            for (int i = 0; i < 8; ++i) { a[i] = alo[i]; a[i + 8] = ahi[i]; }
            // load all four B fragments first -> one clause, one dscnt wait
            v16h bfrag[4];
#pragma unroll
            for (int t = 0; t < 4; ++t)
                bfrag[t] = *(const v16h*)(Bs + (size_t)(t * 16 + lm) * K + kk + half * 16);
#pragma unroll
            for (int t = 0; t < 4; ++t)
                acc[t] = __builtin_amdgcn_wmma_f32_16x16x32_f16(
                    false, a, false, bfrag[t], (short)0, acc[t], false, false);
        }
#pragma unroll
        for (int t = 0; t < 4; ++t) {
            int n = nBase + t * 16 + lm;
            float bv = bias ? bias[n] : 0.f;
#pragma unroll
            for (int r = 0; r < 8; ++r) {
                float y = acc[t][r] + bv;
                if (act == 1) y = fmaxf(y, 0.f);
                int m = m0 + r + half * 8;
                C[(size_t)m * N + n] = (_Float16)y;
            }
        }
    }
}

// ---------------------------------------------------------------------------
// Flash attention: one wave per (batch, head, 16-query tile).
// 32 keys per iteration: 2 score WMMAs, online softmax with shfl_xor row
// reductions inside each 16-lane half, LDS repack P (C->A layout), 2 O WMMAs.
// ---------------------------------------------------------------------------
__global__ void k_flash(const _Float16* __restrict__ qkv, _Float16* __restrict__ out,
                        int nTok, int tokStride,
                        int qBase, int kBase, int vBase, int batchStride,
                        int dh, float scale,
                        int outTokStride, int outBatchStride, int outHeadStride) {
    __shared__ _Float16 pl[16 * 32];
    const int lane = threadIdx.x & 31;
    const int half = lane >> 4;
    const int lm = lane & 15;
    const int qt = blockIdx.x;
    const int h = blockIdx.y;
    const int b = blockIdx.z;
    const int qo = qBase + b * batchStride + h * dh;
    const int ko = kBase + b * batchStride + h * dh;
    const int vo = vBase + b * batchStride + h * dh;
    const int m = qt * 16 + lm;
    const int ndc = (dh + 31) / 32;
    const int ntc = (dh + 15) / 16;

    v16h aq[2];
#pragma unroll
    for (int cc = 0; cc < 2; ++cc) {
        v16h a = {};
        if (cc < ndc) {
            int dBase = cc * 32;
#pragma unroll
            for (int i = 0; i < 16; ++i) {
                int k = dBase + half * 16 + i;
                a[i] = (k < dh) ? qkv[(size_t)m * tokStride + qo + k] : (_Float16)0;
            }
        }
        aq[cc] = a;
    }

    float rowM[8], rowL[8];
#pragma unroll
    for (int r = 0; r < 8; ++r) { rowM[r] = -1e30f; rowL[r] = 0.f; }
    v8f zero = {};
    v8f oc[4];
#pragma unroll
    for (int t = 0; t < 4; ++t) oc[t] = zero;

    for (int j0 = 0; j0 < nTok; j0 += 32) {
        v8f st[2];
#pragma unroll
        for (int t = 0; t < 2; ++t) {
            int key = j0 + t * 16 + lm;
            v8f s = zero;
#pragma unroll
            for (int cc = 0; cc < 2; ++cc) {
                if (cc < ndc) {
                    int dBase = cc * 32 + half * 16;
                    v16h bk = {};
                    if (dBase + 16 <= dh)
                        bk = *(const v16h*)(qkv + (size_t)key * tokStride + ko + dBase);
                    s = __builtin_amdgcn_wmma_f32_16x16x32_f16(
                        false, aq[cc], false, bk, (short)0, s, false, false);
                }
            }
#pragma unroll
            for (int r = 0; r < 8; ++r) s[r] *= scale;
            st[t] = s;
        }
#pragma unroll
        for (int r = 0; r < 8; ++r) {
            float mx = fmaxf(st[0][r], st[1][r]);
            for (int off = 1; off < 16; off <<= 1) mx = fmaxf(mx, __shfl_xor(mx, off, 32));
            float nm = fmaxf(rowM[r], mx);
            float corr = __expf(rowM[r] - nm);
            rowM[r] = nm;
            float p0 = __expf(st[0][r] - nm);
            float p1 = __expf(st[1][r] - nm);
            st[0][r] = p0; st[1][r] = p1;
            float rs = p0 + p1;
            for (int off = 1; off < 16; off <<= 1) rs += __shfl_xor(rs, off, 32);
            rowL[r] = rowL[r] * corr + rs;
#pragma unroll
            for (int nt = 0; nt < 4; ++nt) oc[nt][r] *= corr;
        }
#pragma unroll
        for (int t = 0; t < 2; ++t)
#pragma unroll
            for (int r = 0; r < 8; ++r)
                pl[(r + half * 8) * 32 + t * 16 + lm] = (_Float16)st[t][r];
        __syncthreads();
        v16h pa;
#pragma unroll
        for (int i = 0; i < 16; ++i) {
            int k = half * 16 + i;
            pa[i] = pl[lm * 32 + k];
        }
        __syncthreads();
#pragma unroll
        for (int nt = 0; nt < 4; ++nt) {
            if (nt < ntc) {
                int d = nt * 16 + lm;
                v16h bv = {};
                if (d < dh) {
#pragma unroll
                    for (int i = 0; i < 16; ++i)
                        bv[i] = qkv[(size_t)(j0 + half * 16 + i) * tokStride + vo + d];
                }
                oc[nt] = __builtin_amdgcn_wmma_f32_16x16x32_f16(
                    false, pa, false, bv, (short)0, oc[nt], false, false);
            }
        }
    }
    size_t ob = (size_t)b * outBatchStride + (size_t)h * outHeadStride;
#pragma unroll
    for (int nt = 0; nt < 4; ++nt) {
        if (nt < ntc) {
            int d = nt * 16 + lm;
            if (d < dh) {
#pragma unroll
                for (int r = 0; r < 8; ++r) {
                    int mrow = qt * 16 + r + half * 8;
                    out[(size_t)mrow * outTokStride + ob + d] = (_Float16)(oc[nt][r] / rowL[r]);
                }
            }
        }
    }
}

// ---------------------------------------------------------------------------
// Residual-add + LayerNorm over D=128, optional SiLU. One wave per row.
// ---------------------------------------------------------------------------
__global__ void k_ln(const _Float16* __restrict__ X, const _Float16* __restrict__ R,
                     const float* __restrict__ g, const float* __restrict__ be,
                     _Float16* __restrict__ Y, int M, int silu) {
    const int lane = threadIdx.x & 31;
    const int wave = threadIdx.x >> 5;
    const int row = blockIdx.x * 8 + wave;
    if (row >= M) return;
    const size_t base = (size_t)row * 128;
    float x[4];
#pragma unroll
    for (int i = 0; i < 4; ++i) {
        int d = lane * 4 + i;
        float v = (float)X[base + d];
        if (R) v += (float)R[base + d];
        x[i] = v;
    }
    float s = x[0] + x[1] + x[2] + x[3];
    for (int off = 1; off < 32; off <<= 1) s += __shfl_xor(s, off, 32);
    float mean = s * (1.f / 128.f);
    float vs = 0.f;
#pragma unroll
    for (int i = 0; i < 4; ++i) { float d0 = x[i] - mean; vs += d0 * d0; }
    for (int off = 1; off < 32; off <<= 1) vs += __shfl_xor(vs, off, 32);
    float rstd = __frsqrt_rn(vs * (1.f / 128.f) + 1e-5f);
#pragma unroll
    for (int i = 0; i < 4; ++i) {
        int d = lane * 4 + i;
        float y = (x[i] - mean) * rstd * g[d] + be[d];
        if (silu) y = y / (1.f + __expf(-y));
        Y[base + d] = (_Float16)y;
    }
}

// ---------------------------------------------------------------------------
// Transpose [B,S,128] -> [S,B,128] with pe(b,d) added (reference quirk).
// ---------------------------------------------------------------------------
__global__ void k_pe_transpose(const _Float16* __restrict__ E, _Float16* __restrict__ H) {
    int idx = blockIdx.x * blockDim.x + threadIdx.x;
    if (idx >= 128 * 256 * 128) return;
    int d = idx & 127;
    int row = idx >> 7;
    int b = row & 127;
    int s = row >> 7;
    float v = (float)E[((size_t)b * 256 + s) * 128 + d];
    float freq = __expf((float)(d & ~1) * (-9.2103403720f / 128.f));
    float ang = (float)b * freq;
    v += (d & 1) ? __cosf(ang) : __sinf(ang);
    H[idx] = (_Float16)v;
}

// ---------------------------------------------------------------------------
// Conv1d (SAME) as sum of shifted WMMA GEMMs over D=128, fused ReLU.
// ---------------------------------------------------------------------------
__global__ void k_conv(const _Float16* __restrict__ H,
                       const _Float16* __restrict__ W1_, const float* __restrict__ b1_,
                       const _Float16* __restrict__ W2_, const float* __restrict__ b2_,
                       const _Float16* __restrict__ W3_, const float* __restrict__ b3_,
                       _Float16* __restrict__ Y) {
    const int lane = threadIdx.x & 31;
    const int wave = threadIdx.x >> 5;
    const int mTile = blockIdx.x * 4 + wave;
    const int grp = blockIdx.y;
    const int half = lane >> 4;
    const int lm = lane & 15;
    const int m0 = mTile * 16;
    const int s = m0 >> 7;
    const int ksz = (grp == 0) ? 3 : (grp == 1) ? 5 : 7;
    const _Float16* Wc = (grp == 0) ? W1_ : (grp == 1) ? W2_ : W3_;
    const float* bc = (grp == 0) ? b1_ : (grp == 1) ? b2_ : b3_;
    const int c = ksz / 2;

    v8f zero = {};
    v8f acc[4];
#pragma unroll
    for (int t = 0; t < 4; ++t) acc[t] = zero;

    for (int t = 0; t < ksz; ++t) {
        int ss = s + t - c;
        if (ss < 0 || ss >= 256) continue;
        const _Float16* Arow = H + ((size_t)(m0 + (t - c) * 128) + lm) * 128;
        for (int dc = 0; dc < 128; dc += 32) {
            const _Float16* ap = Arow + dc + half * 16;
            v8h alo = *(const v8h*)(ap);
            v8h ahi = *(const v8h*)(ap + 8);
            v16h a;
#pragma unroll
            for (int i = 0; i < 8; ++i) { a[i] = alo[i]; a[i + 8] = ahi[i]; }
#pragma unroll
            for (int nt = 0; nt < 4; ++nt) {
                int o = nt * 16 + lm;
                v16h bw;
#pragma unroll
                for (int i = 0; i < 16; ++i) {
                    int d = dc + half * 16 + i;
                    bw[i] = Wc[(size_t)o * 128 * ksz + (size_t)d * ksz + t];
                }
                acc[nt] = __builtin_amdgcn_wmma_f32_16x16x32_f16(
                    false, a, false, bw, (short)0, acc[nt], false, false);
            }
        }
    }
#pragma unroll
    for (int nt = 0; nt < 4; ++nt) {
        int o = nt * 16 + lm;
        float bv = bc[o];
#pragma unroll
        for (int r = 0; r < 8; ++r) {
            float y = fmaxf(acc[nt][r] + bv, 0.f);
            int mrow = m0 + r + half * 8;
            Y[(size_t)mrow * 192 + grp * 64 + o] = (_Float16)y;
        }
    }
}

__global__ void k_maxreduce(const _Float16* __restrict__ Y, _Float16* __restrict__ C) {
    int idx = blockIdx.x * blockDim.x + threadIdx.x;
    if (idx >= 128 * 192) return;
    int n = idx % 192, b = idx / 192;
    float mx = -1e30f;
    for (int s = 0; s < 256; ++s)
        mx = fmaxf(mx, (float)Y[((size_t)s * 128 + b) * 192 + n]);
    C[(size_t)b * 192 + n] = (_Float16)mx;
}

__global__ void k_head(const _Float16* __restrict__ Z, const float* __restrict__ vW,
                       const float* __restrict__ vb, const float* __restrict__ aW,
                       const float* __restrict__ ab, float* __restrict__ out) {
    int b = blockIdx.x, j = threadIdx.x;
    const _Float16* z = Z + (size_t)b * 64;
    float adv = 0.f;
    for (int k = 0; k < 64; ++k) adv += (float)z[k] * aW[j * 64 + k];
    adv += ab[j];
    float val = 0.f;
    for (int k = j; k < 64; k += 32) val += (float)z[k] * vW[k];
    for (int off = 1; off < 32; off <<= 1) val += __shfl_xor(val, off, 32);
    val += vb[0];
    float mean = adv;
    for (int off = 1; off < 32; off <<= 1) mean += __shfl_xor(mean, off, 32);
    mean *= (1.f / 32.f);
    out[(size_t)b * 32 + j] = val + adv - mean;
}

// ---------------------------------------------------------------------------
// Host orchestration. Input order: x, cluster_ids, history, emb_W, emb_b,
//   3x[Wi,bi,Wo,bo,W1,b1,W2,b2,g1,be1,g2,be2],
//   c1W,c1b,c2W,c2b,c3W,c3b, aWi,abi,aWo,abo, clW1,clb1,clg,clbe, clW2,clb2,
//   vW,vb,adW,adb  (61 inputs).
// ---------------------------------------------------------------------------
extern "C" void kernel_launch(void* const* d_in, const int* in_sizes, int n_in,
                              void* d_out, int out_size, void* d_ws, size_t ws_size,
                              hipStream_t stream) {
    (void)in_sizes; (void)out_size; (void)ws_size;
    if (n_in < 61) return;
    const int M = 256 * 128;

    const float* hist = (const float*)d_in[2];
    const float* embW = (const float*)d_in[3];
    const float* embB = (const float*)d_in[4];
    auto L = [&](int l, int j) { return (const float*)d_in[5 + l * 12 + j]; };
    const float* c1W = (const float*)d_in[41]; const float* c1b = (const float*)d_in[42];
    const float* c2W = (const float*)d_in[43]; const float* c2b = (const float*)d_in[44];
    const float* c3W = (const float*)d_in[45]; const float* c3b = (const float*)d_in[46];
    const float* aWi = (const float*)d_in[47]; const float* abi = (const float*)d_in[48];
    const float* aWo = (const float*)d_in[49]; const float* abo = (const float*)d_in[50];
    const float* clW1 = (const float*)d_in[51]; const float* clb1 = (const float*)d_in[52];
    const float* clg = (const float*)d_in[53];  const float* clbe = (const float*)d_in[54];
    const float* clW2 = (const float*)d_in[55]; const float* clb2 = (const float*)d_in[56];
    const float* vW = (const float*)d_in[57];   const float* vb = (const float*)d_in[58];
    const float* adW = (const float*)d_in[59];  const float* adb = (const float*)d_in[60];

    char* wsb = (char*)d_ws;
    size_t off = 0;
    auto alloc = [&](size_t n) -> _Float16* {
        _Float16* p = (_Float16*)(wsb + off);
        off = (off + n * sizeof(_Float16) + 255) & ~(size_t)255;
        return p;
    };
    _Float16* HIST16 = alloc((size_t)M * 64);
    _Float16* EMBW16 = alloc(128 * 64);
    _Float16* WI16[3], *WO16[3], *W116[3], *W216[3];
    for (int l = 0; l < 3; ++l) {
        WI16[l] = alloc(384 * 128); WO16[l] = alloc(128 * 128);
        W116[l] = alloc(512 * 128); W216[l] = alloc(128 * 512);
    }
    _Float16* C1W16 = alloc(64 * 128 * 3);
    _Float16* C2W16 = alloc(64 * 128 * 5);
    _Float16* C3W16 = alloc(64 * 128 * 7);
    _Float16* AWI16 = alloc(576 * 192);
    _Float16* AWO16 = alloc(192 * 192);
    _Float16* CLW116 = alloc(128 * 192);
    _Float16* CLW216 = alloc(64 * 128);
    _Float16* Hbuf = alloc((size_t)M * 128);
    _Float16* ATT = alloc((size_t)M * 128);
    _Float16* SHARED = alloc((size_t)M * 512);
    _Float16* SQKV = alloc(128 * 576);
    _Float16* COMB = alloc(128 * 192);
    _Float16* APROJ = alloc(128 * 192);
    _Float16* Z = alloc(128 * 128);
    _Float16* Z2 = alloc(128 * 64);

    auto cvt = [&](const float* src, _Float16* dst, int n) {
        k_cvt<<<(n + 255) / 256, 256, 0, stream>>>(src, dst, n);
    };
    auto gemm = [&](const _Float16* A, const _Float16* W, const float* bias, _Float16* C,
                    int Mm, int N, int K, int lda, int act) {
        dim3 g((Mm / 16 + 3) / 4, N / 64);
        size_t smem = (size_t)K * 64 * sizeof(_Float16);  // B panel
        k_gemm<<<g, 128, smem, stream>>>(A, W, bias, C, Mm, N, K, lda, act);
    };

    cvt(hist, HIST16, M * 64);
    cvt(embW, EMBW16, 128 * 64);
    for (int l = 0; l < 3; ++l) {
        cvt(L(l, 0), WI16[l], 384 * 128);
        cvt(L(l, 2), WO16[l], 128 * 128);
        cvt(L(l, 4), W116[l], 512 * 128);
        cvt(L(l, 6), W216[l], 128 * 512);
    }
    cvt(c1W, C1W16, 64 * 128 * 3);
    cvt(c2W, C2W16, 64 * 128 * 5);
    cvt(c3W, C3W16, 64 * 128 * 7);
    cvt(aWi, AWI16, 576 * 192);
    cvt(aWo, AWO16, 192 * 192);
    cvt(clW1, CLW116, 128 * 192);
    cvt(clW2, CLW216, 64 * 128);

    // embedding + transpose + positional encoding
    gemm(HIST16, EMBW16, embB, SHARED, M, 128, 64, 64, 0);
    k_pe_transpose<<<(M * 128 + 255) / 256, 256, 0, stream>>>(SHARED, Hbuf);

    // 3 transformer encoder layers
    for (int l = 0; l < 3; ++l) {
        gemm(Hbuf, WI16[l], L(l, 1), SHARED, M, 384, 128, 128, 0);
        k_flash<<<dim3(16, 4, 128), 32, 0, stream>>>(
            SHARED, ATT, 256, 49152, 0, 128, 256, 384,
            32, 0.17677669529f, 16384, 128, 32);
        gemm(ATT, WO16[l], L(l, 3), SHARED, M, 128, 128, 128, 0);
        k_ln<<<M / 8, 256, 0, stream>>>(Hbuf, SHARED, L(l, 8), L(l, 9), Hbuf, M, 0);
        gemm(Hbuf, W116[l], L(l, 5), SHARED, M, 512, 128, 128, 1);
        gemm(SHARED, W216[l], L(l, 7), ATT, M, 128, 512, 512, 0);
        k_ln<<<M / 8, 256, 0, stream>>>(Hbuf, ATT, L(l, 10), L(l, 11), Hbuf, M, 0);
    }

    // conv heads + relu + max over sequence
    k_conv<<<dim3(512, 3), 128, 0, stream>>>(Hbuf, C1W16, c1b, C2W16, c2b, C3W16, c3b, SHARED);
    k_maxreduce<<<(128 * 192 + 255) / 256, 256, 0, stream>>>(SHARED, COMB);

    // "small attention": seq length is 1 -> softmax==1 -> output = V proj
    gemm(COMB, AWI16, abi, SQKV, 128, 576, 192, 192, 0);
    gemm(SQKV + 384, AWO16, abo, APROJ, 128, 192, 192, 576, 0);

    // cluster MLP + dueling head
    gemm(APROJ, CLW116, clb1, Z, 128, 128, 192, 192, 0);
    k_ln<<<16, 256, 0, stream>>>(Z, nullptr, clg, clbe, Z, 128, 1);
    gemm(Z, CLW216, clb2, Z2, 128, 64, 128, 128, 0);
    k_head<<<128, 32, 0, stream>>>(Z2, vW, vb, adW, adb, (float*)d_out);
}
